// Linear_72816875536475
// MI455X (gfx1250) — matrix operand order repaired and checked by reading, not compiled, not run on hardware
//
#include <hip/hip_runtime.h>

typedef __attribute__((ext_vector_type(16))) _Float16 v16h;
typedef __attribute__((ext_vector_type(8)))  _Float16 v8h;
typedef __attribute__((ext_vector_type(8)))  float    v8f;
typedef __attribute__((ext_vector_type(4)))  int      v4i;

#define IN_DIM   8192
#define OUT_DIM  8192
#define TOK      16
#define QBLOCK   32
#define KSPLIT   4
#define KPW      (IN_DIM / KSPLIT)      // 2048 -> 64 WMMA steps per wave
#define NTILES   (OUT_DIM / 16)         // 512 column tiles
#define TPB      8                      // tiles (waves) per 256-thread block

// ---------------------------------------------------------------------------
// Prep: convert activations f32 -> f16 once (A matrix reused by 512 tiles).
// ---------------------------------------------------------------------------
__global__ void x_to_f16(const float* __restrict__ x, _Float16* __restrict__ xh) {
    int i = blockIdx.x * blockDim.x + threadIdx.x;
    if (i < TOK * IN_DIM) xh[i] = (_Float16)x[i];
}

// ---------------------------------------------------------------------------
// Main: dequant-on-the-fly WMMA GEMM.  One wave owns a 16(tok) x 16(out) tile
// for one K-slice.  K=32 per WMMA == one ggml quant block, so the per-block
// scale is uniform inside each B fragment.
// ---------------------------------------------------------------------------
__global__ __launch_bounds__(256) void qgemm_wmma(
        const int*     __restrict__ qw,      // [OUT, IN] int32 codes
        const float*   __restrict__ scales,  // [OUT, IN/32]
        const _Float16* __restrict__ xh,     // [TOK, IN] f16
        float*         __restrict__ partial) // [KSPLIT, TOK, OUT]
{
    const int wave = threadIdx.x >> 5;
    const int lane = threadIdx.x & 31;
    const int tile = blockIdx.x * TPB + wave;        // 0..511
    const int ks   = blockIdx.y;                     // 0..KSPLIT-1
    const int nlo  = lane & 15;                      // column-in-tile / row-in-tile
    const int half = lane >> 4;                      // lane group 0/1
    const int col  = (tile << 4) + nlo;              // output column

    const _Float16* aRow = xh + (size_t)nlo * IN_DIM;            // token row = nlo
    const int*      qRow = qw + (size_t)col * IN_DIM;
    const float*    sRow = scales + (size_t)col * (IN_DIM / QBLOCK);

    v8f acc = {};
    const int k0 = ks * KPW;

    #pragma unroll 2
    for (int k = k0; k < k0 + KPW; k += QBLOCK) {
        // ---- A fragment (contiguous k order, not the ISA table order):
        //      elems 0..15 -> K = k + 16*half + e (matches B)
        v8h a0 = *(const v8h*)(aRow + k + 16 * half);
        v8h a1 = *(const v8h*)(aRow + k + 16 * half + 8);
        v16h a;
        #pragma unroll
        for (int e = 0; e < 8; ++e) { a[e] = a0[e]; a[8 + e] = a1[e]; }

        // ---- B fragment: lane holds column `col`, K = k + 16*half + e
        //      (16 consecutive codes -> 64 contiguous bytes, streamed NT)
        const v4i* qp = (const v4i*)(qRow + k + 16 * half);
        v4i c0 = __builtin_nontemporal_load(qp + 0);
        v4i c1 = __builtin_nontemporal_load(qp + 1);
        v4i c2 = __builtin_nontemporal_load(qp + 2);
        v4i c3 = __builtin_nontemporal_load(qp + 3);
        const float s = sRow[k >> 5];   // one scale per (col, block)

        v16h b;
        #pragma unroll
        for (int e = 0; e < 4; ++e) {
            b[e]      = (_Float16)((float)(c0[e] - 127) * s);
            b[4 + e]  = (_Float16)((float)(c1[e] - 127) * s);
            b[8 + e]  = (_Float16)((float)(c2[e] - 127) * s);
            b[12 + e] = (_Float16)((float)(c3[e] - 127) * s);
        }

        // D(f32) += A(f16 16x32) * B(f16 32x16)
        acc = __builtin_amdgcn_wmma_f32_16x16x32_f16(
                  false, a, false, b, (short)0, acc, false, false);
    }

    // ---- store partial tile.  C/D layout: VGPR r, lanes 0-15 -> M=r, N=lane;
    //      lanes 16-31 -> M=8+r, N=lane-16.
    float* p = partial + ((size_t)ks * TOK + 8 * half) * OUT_DIM + col;
    #pragma unroll
    for (int r = 0; r < 8; ++r) p[(size_t)r * OUT_DIM] = acc[r];
}

// ---------------------------------------------------------------------------
// Deterministic reduction of K-slice partials + bias.
// ---------------------------------------------------------------------------
__global__ void reduce_bias(const float* __restrict__ partial,
                            const float* __restrict__ bias,
                            float* __restrict__ out) {
    int i = blockIdx.x * blockDim.x + threadIdx.x;
    if (i >= TOK * OUT_DIM) return;
    float v = bias[i & (OUT_DIM - 1)];
    #pragma unroll
    for (int s = 0; s < KSPLIT; ++s) v += partial[(size_t)s * TOK * OUT_DIM + i];
    out[i] = v;
}

extern "C" void kernel_launch(void* const* d_in, const int* in_sizes, int n_in,
                              void* d_out, int out_size, void* d_ws, size_t ws_size,
                              hipStream_t stream) {
    const float* x      = (const float*)d_in[0];
    const int*   qw     = (const int*)d_in[1];
    const float* scales = (const float*)d_in[2];
    const float* bias   = (const float*)d_in[3];
    float*       out    = (float*)d_out;

    _Float16* xh      = (_Float16*)d_ws;
    float*    partial = (float*)((char*)d_ws + (size_t)TOK * IN_DIM * sizeof(_Float16));

    x_to_f16<<<(TOK * IN_DIM + 255) / 256, 256, 0, stream>>>(x, xh);

    dim3 grid(NTILES / TPB, KSPLIT);
    qgemm_wmma<<<grid, 256, 0, stream>>>(qw, scales, xh, partial);

    reduce_bias<<<(TOK * OUT_DIM + 255) / 256, 256, 0, stream>>>(partial, bias, out);
}
